// Dec_module_74285754351968
// MI455X (gfx1250) — matrix operand order repaired and checked by reading, not compiled, not run on hardware
//
#include <hip/hip_runtime.h>
#include <math.h>

// ---------------------------------------------------------------------------
// MI455X (gfx1250) implementation.
//
// fd_split's double roll cancels -> high=|x|, low=0. fem_b sees an all-zero
// input and is batch-invariant -> computed once at B=1 and broadcast.
//
// 3x3 convs: implicit GEMM on V_WMMA_F32_16X16X32_BF16 using a 3-term
// split-precision scheme (A,B decomposed to bf16 hi+lo; Ah*Bh+Ah*Bl+Al*Bh,
// fp32 accumulate) -> ~16-bit mantissa accuracy at bf16 matrix-pipe rate.
// Weights are repacked to a contiguous-k A-fragment layout (1 contiguous 32B load
// per lane); activations are staged per-block into LDS as bf16 hi/lo with
// halo zeros, so B-fragments are contiguous 32B ds reads.
//
// 1x1 expert tail (~7% of FLOPs) stays on exact V_WMMA_F32_16X16X4_F32 with
// fused elementwise product, bias, gate scale and accumulation.
// ---------------------------------------------------------------------------

typedef float   v2f   __attribute__((ext_vector_type(2)));
typedef float   v8f   __attribute__((ext_vector_type(8)));
typedef __bf16  v16bf __attribute__((ext_vector_type(16)));
typedef __bf16  bf16;

static constexpr int  CIN  = 128;
static constexpr int  HH   = 128;
static constexpr int  WW   = 128;
static constexpr int  HW   = HH * WW;           // 16384
static constexpr long CHW  = (long)CIN * HW;    // 2,097,152
static constexpr long NCHW = 8L * CHW;          // 16,777,216
static constexpr int  LDSC = 136;               // padded channel stride (bank spread)

// ---------------------------------------------------------------------------
// Repack OIHW fp32 3x3 weights into bf16 hi/lo WMMA A-fragments:
// dims [OC/16][tap=9][cb=4][lane=32][j=16]; lane/j use this code's contiguous-k 16-bit
// 16x32 A layout: K(j,lane) = j + 16*(lane>=16), M = lane&15.
// ---------------------------------------------------------------------------
__global__ __launch_bounds__(256) void repack_w3x3(
    const float* __restrict__ w, bf16* __restrict__ whi,
    bf16* __restrict__ wlo, int OC) {
  const long i = (long)blockIdx.x * 256 + threadIdx.x;   // over OC*1152
  if (i >= (long)OC * 1152) return;
  const int j    = (int)(i & 15);
  long t = i >> 4;
  const int lane = (int)(t & 31); t >>= 5;
  const int cb   = (int)(t & 3);  t >>= 2;
  const int tap  = (int)(t % 9);
  const long octile = t / 9;
  const int oc = (int)(octile * 16 + (lane & 15));
  const int K  = j + ((lane >> 4) << 4);
  const int c  = cb * 32 + K;
  const float v = w[(long)oc * (CIN * 9) + (long)c * 9 + tap];
  const bf16 hi = (bf16)v;
  whi[i] = hi;
  wlo[i] = (bf16)(v - (float)hi);
}

// ---------------------------------------------------------------------------
// 3x3 conv, implicit GEMM via 3-term split bf16 WMMA (16x16x32).
// grid.x = (W/16)*H pixel tiles, grid.y = batch, grid.z = OC/128.
// block = (32,8): 8 waves split 128 out-channels and share one LDS input tile.
// TAKE_ABS fuses |x| from fd_split into the staging load of conv1_a.
// Channel oc<128 -> out0, oc>=128 -> out1 (xh / k split of conv1).
// ---------------------------------------------------------------------------
template <bool TAKE_ABS>
__global__ __launch_bounds__(256) void conv3x3_wmma_bf16(
    const float* __restrict__ in, const bf16* __restrict__ whi,
    const bf16* __restrict__ wlo, const float* __restrict__ bias,
    float* __restrict__ out0, float* __restrict__ out1, int OC) {
  __shared__ bf16 sHi[54 * LDSC];
  __shared__ bf16 sLo[54 * LDSC];

  const int h  = blockIdx.x >> 3;                 // W/16 == 8 tiles per row
  const int w0 = (blockIdx.x & 7) * 16;
  const int b  = blockIdx.y;
  const float* __restrict__ inB = in + (long)b * CHW;

  // ---- stage 3x18x128 halo tile, split to bf16 hi/lo, zeros at borders ----
  const int tid = threadIdx.y * 32 + threadIdx.x;
  for (int idx = tid; idx < 54 * CIN; idx += 256) {
    const int pos = idx % 54;                     // row*18 + col
    const int c   = idx / 54;
    const int row = pos / 18, col = pos % 18;
    const int hh = h - 1 + row;
    const int wc = w0 - 1 + col;
    float v = 0.0f;
    if (hh >= 0 && hh < HH && wc >= 0 && wc < WW)
      v = inB[(long)c * HW + (long)hh * WW + wc];
    if (TAKE_ABS) v = fabsf(v);
    const bf16 hi = (bf16)v;
    sHi[pos * LDSC + c] = hi;
    sLo[pos * LDSC + c] = (bf16)(v - (float)hi);
  }
  __syncthreads();

  const int lane   = threadIdx.x;
  const int oc0    = (blockIdx.z * 8 + threadIdx.y) * 16;
  const int ncol   = lane & 15;
  const int half16 = (lane >> 4) << 4;            // B K-halves: lanes>=16 -> K+16
  const long octile = oc0 >> 4;

  v8f acc = {};
#pragma unroll
  for (int tap = 0; tap < 9; ++tap) {
    const int pos = (tap / 3) * 18 + ncol + (tap % 3);   // halo-shifted position
    const bf16* __restrict__ bh = &sHi[pos * LDSC + half16];
    const bf16* __restrict__ bl = &sLo[pos * LDSC + half16];
#pragma unroll
    for (int cb = 0; cb < 4; ++cb) {
      const long abase = (((octile * 9 + tap) * 4 + cb) << 9) + lane * 16;
      const v16bf Ahi = *(const v16bf*)(whi + abase);
      const v16bf Alo = *(const v16bf*)(wlo + abase);
      const v16bf Bhi = *(const v16bf*)(bh + cb * 32);
      const v16bf Blo = *(const v16bf*)(bl + cb * 32);
      acc = __builtin_amdgcn_wmma_f32_16x16x32_bf16(
          false, Ahi, false, Bhi, (short)0, acc, false, false);
      acc = __builtin_amdgcn_wmma_f32_16x16x32_bf16(
          false, Ahi, false, Blo, (short)0, acc, false, false);
      acc = __builtin_amdgcn_wmma_f32_16x16x32_bf16(
          false, Alo, false, Bhi, (short)0, acc, false, false);
    }
  }

  // C/D: VGPR r -> M = r (lanes 0-15) or 8+r (lanes 16-31); N = lane&15.
  const int mhi = (lane >> 4) << 3;
#pragma unroll
  for (int r = 0; r < 8; ++r) {
    const int oc = oc0 + r + mhi;
    const float v = acc[r] + bias[oc];
    float* __restrict__ op = (oc < CIN) ? out0 : out1;
    const int ocs = oc & (CIN - 1);
    op[((long)b * CIN + ocs) * HW + (long)h * WW + w0 + ncol] = v;
  }
}

// ---------------------------------------------------------------------------
// Expert tail: acc_out += expw[b,e] * ( (a.*b) @ w3^T + b3 )  (1x1 conv GEMM)
// Exact fp32 via V_WMMA_F32_16X16X4_F32; fuses product/bias/gate/accumulate.
// Skips the whole block when the top-k mask zeroed this expert.
// ---------------------------------------------------------------------------
__global__ __launch_bounds__(256) void conv1x1_moe_wmma(
    const float* __restrict__ aB, const float* __restrict__ bB,
    const float* __restrict__ w3, const float* __restrict__ b3,
    const float* __restrict__ expw, int e, float* __restrict__ acc_out) {
  const int b = blockIdx.y;
  const float scale = expw[b * 3 + e];
  if (scale == 0.0f) return;                      // masked-out expert

  const int lane = threadIdx.x;
  const int oc0  = threadIdx.y * 16;              // 8 waves cover OC=128
  const int p0   = blockIdx.x * 16;
  const int mrow = lane & 15;
  const int koff = (lane >> 4) << 1;
  const int ncol = lane & 15;

  const float* __restrict__ aP = aB + (long)b * CHW + p0 + ncol;
  const float* __restrict__ bP = bB + (long)b * CHW + p0 + ncol;
  const float* __restrict__ wP = w3 + (long)(oc0 + mrow) * CIN;

  v8f acc = {};
  for (int c = 0; c < CIN; c += 4) {
    const int ka = c + koff;
    v2f A, B;
    A.x = wP[ka];
    A.y = wP[ka + 1];
    B.x = aP[(long)ka * HW] * bP[(long)ka * HW];
    B.y = aP[(long)(ka + 1) * HW] * bP[(long)(ka + 1) * HW];
    acc = __builtin_amdgcn_wmma_f32_16x16x4_f32(
        false, A, false, B, (short)0, acc, false, false);
  }

  const int mhi = (lane >> 4) << 3;
#pragma unroll
  for (int r = 0; r < 8; ++r) {
    const int oc = oc0 + r + mhi;
    const long o = ((long)b * CIN + oc) * HW + p0 + ncol;
    acc_out[o] += scale * (acc[r] + b3[oc]);
  }
}

// ---------------------------------------------------------------------------
__global__ __launch_bounds__(256) void mean_hw_kernel(
    const float* __restrict__ xh, float* __restrict__ mean) {
  __shared__ float sb[256];
  const float* __restrict__ p = xh + (long)blockIdx.x * HW;
  float s = 0.0f;
  for (int i = threadIdx.x; i < HW; i += 256) s += p[i];
  sb[threadIdx.x] = s;
  __syncthreads();
  for (int st = 128; st > 0; st >>= 1) {
    if (threadIdx.x < st) sb[threadIdx.x] += sb[threadIdx.x + st];
    __syncthreads();
  }
  if (threadIdx.x == 0) mean[blockIdx.x] = sb[0] * (1.0f / HW);
}

__global__ void gate_kernel(const float* __restrict__ mean,
                            const float* __restrict__ gw,
                            float* __restrict__ expw, int Bn) {
  const int b = threadIdx.x;
  if (b >= Bn) return;
  float lg[3];
  for (int e = 0; e < 3; ++e) {
    float s = 0.0f;
    for (int c = 0; c < CIN; ++c) s += mean[b * CIN + c] * gw[e * CIN + c];
    lg[e] = s;
  }
  const float mx = fmaxf(lg[0], fmaxf(lg[1], lg[2]));
  float ex[3], den = 0.0f;
  for (int e = 0; e < 3; ++e) { ex[e] = expf(lg[e] - mx); den += ex[e]; }
  float w[3];
  for (int e = 0; e < 3; ++e) w[e] = ex[e] / den;
  int mi = 0;                                     // drop the min (ties -> last)
  for (int j = 1; j < 3; ++j) if (w[j] <= w[mi]) mi = j;
  for (int e = 0; e < 3; ++e) expw[b * 3 + e] = (e == mi) ? 0.0f : w[e];
}

__global__ __launch_bounds__(256) void zero_kernel(float* __restrict__ p, long n) {
  const long i = (long)blockIdx.x * blockDim.x + threadIdx.x;
  if (i < n) p[i] = 0.0f;
}

__global__ __launch_bounds__(256) void final_kernel(
    const float* __restrict__ xh_a, const float* __restrict__ ea,
    const float* __restrict__ xh_b, const float* __restrict__ eb,
    const float* __restrict__ x, float* __restrict__ out) {
  const long i = (long)blockIdx.x * blockDim.x + threadIdx.x;
  if (i < NCHW) {
    const long r = i % CHW;
    out[i] = xh_a[i] + ea[i] + xh_b[r] + eb[r] + x[i];
  }
}

// ---------------------------------------------------------------------------
extern "C" void kernel_launch(void* const* d_in, const int* in_sizes, int n_in,
                              void* d_out, int out_size, void* d_ws, size_t ws_size,
                              hipStream_t stream) {
  (void)in_sizes; (void)n_in; (void)out_size; (void)ws_size;
  const float* x      = (const float*)d_in[0];
  const float* a_c1w  = (const float*)d_in[1];
  const float* a_c1b  = (const float*)d_in[2];
  const float* a_gw   = (const float*)d_in[3];
  const float* a_ew1  = (const float*)d_in[4];
  const float* a_eb1  = (const float*)d_in[5];
  const float* a_ew2  = (const float*)d_in[6];
  const float* a_eb2  = (const float*)d_in[7];
  const float* a_ew3  = (const float*)d_in[8];
  const float* a_eb3  = (const float*)d_in[9];
  const float* b_c1w  = (const float*)d_in[10];
  const float* b_c1b  = (const float*)d_in[11];
  const float* b_gw   = (const float*)d_in[12];
  const float* b_ew1  = (const float*)d_in[13];
  const float* b_eb1  = (const float*)d_in[14];
  const float* b_ew2  = (const float*)d_in[15];
  const float* b_eb2  = (const float*)d_in[16];
  const float* b_ew3  = (const float*)d_in[17];
  const float* b_eb3  = (const float*)d_in[18];

  // --- bf16 weight-pack scratch (reused, stream-ordered): max OC=256 conv ---
  const long WMAX = 256L * CIN * 9;               // 294,912 elements
  bf16* whi = (bf16*)d_ws;
  bf16* wlo = whi + WMAX;
  float* fb = (float*)(wlo + WMAX);               // 1,179,648 B in -> aligned

  // --- fp32 workspace (~370 MB) ---
  float* xh_a   = fb;                // NCHW
  float* k_a    = xh_a  + NCHW;      // NCHW
  float* aBuf   = k_a   + NCHW;      // NCHW (reused by fem_b at B=1)
  float* bBuf   = aBuf  + NCHW;      // NCHW
  float* eacc_a = bBuf  + NCHW;      // NCHW
  float* zin_b  = eacc_a + NCHW;     // CHW (all-zero fem_b input)
  float* xh_b   = zin_b + CHW;       // CHW
  float* k_b    = xh_b  + CHW;       // CHW
  float* eacc_b = k_b   + CHW;       // CHW
  float* meanA  = eacc_b + CHW;      // 8*128
  float* expwA  = meanA + 1024;      // 24
  float* meanB  = expwA + 32;        // 128
  float* expwB  = meanB + 128;       // 3

  const dim3 blk(32, 8);
  const long EW1 = (long)CIN * CIN * 9;
  const long EW3 = (long)CIN * CIN;
  const int RG1 = (int)((256L * 1152 + 255) / 256);   // repack grid, OC=256
  const int RG2 = (int)((128L * 1152 + 255) / 256);   // repack grid, OC=128

  // ---------------- fem A on |x| (B=8) ----------------
  repack_w3x3<<<RG1, 256, 0, stream>>>(a_c1w, whi, wlo, 256);
  conv3x3_wmma_bf16<true><<<dim3(1024, 8, 2), blk, 0, stream>>>(
      x, whi, wlo, a_c1b, xh_a, k_a, 256);
  mean_hw_kernel<<<1024, 256, 0, stream>>>(xh_a, meanA);
  gate_kernel<<<1, 32, 0, stream>>>(meanA, a_gw, expwA, 8);
  zero_kernel<<<(NCHW + 255) / 256, 256, 0, stream>>>(eacc_a, NCHW);
  for (int e = 0; e < 3; ++e) {
    repack_w3x3<<<RG2, 256, 0, stream>>>(a_ew1 + e * EW1, whi, wlo, 128);
    conv3x3_wmma_bf16<false><<<dim3(1024, 8, 1), blk, 0, stream>>>(
        xh_a, whi, wlo, a_eb1 + e * CIN, aBuf, aBuf, 128);
    repack_w3x3<<<RG2, 256, 0, stream>>>(a_ew2 + e * EW1, whi, wlo, 128);
    conv3x3_wmma_bf16<false><<<dim3(1024, 8, 1), blk, 0, stream>>>(
        k_a, whi, wlo, a_eb2 + e * CIN, bBuf, bBuf, 128);
    conv1x1_moe_wmma<<<dim3(1024, 8), blk, 0, stream>>>(
        aBuf, bBuf, a_ew3 + e * EW3, a_eb3 + e * CIN, expwA, e, eacc_a);
  }

  // ---------------- fem B on 0 (batch-invariant, B=1) ----------------
  zero_kernel<<<(CHW + 255) / 256, 256, 0, stream>>>(zin_b, CHW);
  zero_kernel<<<(CHW + 255) / 256, 256, 0, stream>>>(eacc_b, CHW);
  repack_w3x3<<<RG1, 256, 0, stream>>>(b_c1w, whi, wlo, 256);
  conv3x3_wmma_bf16<false><<<dim3(1024, 1, 2), blk, 0, stream>>>(
      zin_b, whi, wlo, b_c1b, xh_b, k_b, 256);
  mean_hw_kernel<<<128, 256, 0, stream>>>(xh_b, meanB);
  gate_kernel<<<1, 32, 0, stream>>>(meanB, b_gw, expwB, 1);
  for (int e = 0; e < 3; ++e) {
    repack_w3x3<<<RG2, 256, 0, stream>>>(b_ew1 + e * EW1, whi, wlo, 128);
    conv3x3_wmma_bf16<false><<<dim3(1024, 1, 1), blk, 0, stream>>>(
        xh_b, whi, wlo, b_eb1 + e * CIN, aBuf, aBuf, 128);
    repack_w3x3<<<RG2, 256, 0, stream>>>(b_ew2 + e * EW1, whi, wlo, 128);
    conv3x3_wmma_bf16<false><<<dim3(1024, 1, 1), blk, 0, stream>>>(
        k_b, whi, wlo, b_eb2 + e * CIN, bBuf, bBuf, 128);
    conv1x1_moe_wmma<<<dim3(1024, 1), blk, 0, stream>>>(
        aBuf, bBuf, b_ew3 + e * EW3, b_eb3 + e * CIN, expwB, e, eacc_b);
  }

  // ---------------- combine ----------------
  final_kernel<<<(NCHW + 255) / 256, 256, 0, stream>>>(
      xh_a, eacc_a, xh_b, eacc_b, x, (float*)d_out);
}
